// MultiHeadAttention_25348896981022
// MI455X (gfx1250) — matrix operand order repaired and checked by reading, not compiled, not run on hardware
//
#include <hip/hip_runtime.h>
#include <hip/hip_bf16.h>

typedef _Float16 half_t;
typedef __attribute__((ext_vector_type(16))) _Float16 v16h;
typedef __attribute__((ext_vector_type(8)))  _Float16 v8h;
typedef __attribute__((ext_vector_type(4)))  _Float16 v4h;
typedef __attribute__((ext_vector_type(8)))  float    v8f;
typedef __attribute__((ext_vector_type(4)))  int      v4i;

#define DIM   1024
#define NHEAD 16
#define HD    64
#define BATCH 2
#define SEQ   2048
#define MTOT  (BATCH*SEQ)   // 4096

#if __has_builtin(__builtin_amdgcn_global_load_async_to_lds_b128)
#define HAVE_ASYNC_LDS 1
#else
#define HAVE_ASYNC_LDS 0
#endif

union AFrag { v16h v; v8h h[2]; };

__device__ __forceinline__ v8f wmma16(const AFrag& a, const AFrag& b, v8f c) {
    return __builtin_amdgcn_wmma_f32_16x16x32_f16(
        false, a.v, false, b.v, (short)0, c, false, false);
}

#if HAVE_ASYNC_LDS
__device__ __forceinline__ void async_copy16(const half_t* g, half_t* l) {
    __builtin_amdgcn_global_load_async_to_lds_b128(
        (__attribute__((address_space(1))) v4i*)g,
        (__attribute__((address_space(3))) v4i*)l,
        0, 0);
}
__device__ __forceinline__ void async_wait0() {
#if __has_builtin(__builtin_amdgcn_s_wait_asynccnt)
    __builtin_amdgcn_s_wait_asynccnt(0);
#else
    asm volatile("s_wait_asynccnt 0x0" ::: "memory");
#endif
}
#endif

// ---------------------------------------------------------------- convert ---
__global__ void cvt_f32_to_f16(const float* __restrict__ in,
                               half_t* __restrict__ out, int n4) {
    int i = blockIdx.x * blockDim.x + threadIdx.x;
    if (i < n4) {
        float4 f = ((const float4*)in)[i];
        v4h o;
        o.x = (half_t)f.x; o.y = (half_t)f.y;
        o.z = (half_t)f.z; o.w = (half_t)f.w;
        *(v4h*)(out + 4 * (size_t)i) = o;
    }
}

// ------------------------------------------------------------------- GEMM ---
// C[M,N] = A[M,K] @ Bt[N,K]^T + bias[N]
// mode 0: f16 out, scatter to [B,H,S,HD]     (Q and K)
// mode 1: f16 out, scatter to [B,H,HD,S]     (V transposed)
// mode 2: f32 out, row-major [M,N]           (final projection)
__global__ __launch_bounds__(256) void gemm_nt(
    const half_t* __restrict__ A, const half_t* __restrict__ Bt,
    const float* __restrict__ bias, void* __restrict__ outp,
    int M, int N, int K, int mode)
{
    __shared__ half_t As[128 * 40];  // 128 rows x 32 halfs, stride 40
    __shared__ half_t Bs[64 * 40];   // 64 rows  x 32 halfs, stride 40

    const int tid  = threadIdx.x;
    const int lane = tid & 31, wave = tid >> 5;
    const int g    = lane >> 4, nl = lane & 15;
    const int wm   = wave & 3,  wn = wave >> 2;        // 4x2 wave grid
    const int mb   = blockIdx.y * 128, nb = blockIdx.x * 64;

    v8f acc[2][2] = {};

    const int arow = tid >> 1, achunk = (tid & 1) * 16;  // 16 halfs per thread
    const int brow = tid >> 2, bchunk = (tid & 3) * 8;   //  8 halfs per thread

    for (int k0 = 0; k0 < K; k0 += 32) {
        const half_t* ag = A + (size_t)(mb + arow) * K + k0 + achunk;
        const half_t* bg = Bt + (size_t)(nb + brow) * K + k0 + bchunk;
#if HAVE_ASYNC_LDS
        // gfx1250 async Global->LDS DMA (ASYNCcnt), no VGPR round-trip
        async_copy16(ag,     As + arow * 40 + achunk);
        async_copy16(ag + 8, As + arow * 40 + achunk + 8);
        async_copy16(bg,     Bs + brow * 40 + bchunk);
#else
        *(v8h*)(As + arow * 40 + achunk)     = *(const v8h*)(ag);
        *(v8h*)(As + arow * 40 + achunk + 8) = *(const v8h*)(ag + 8);
        *(v8h*)(Bs + brow * 40 + bchunk)     = *(const v8h*)(bg);
#endif
        if (k0 + 32 < K) {                   // gfx1250 global_prefetch_b8
            __builtin_prefetch(ag + 32, 0, 3);
            __builtin_prefetch(bg + 32, 0, 3);
        }
#if HAVE_ASYNC_LDS
        async_wait0();
#endif
        __syncthreads();

        AFrag a[2];
        #pragma unroll
        for (int mt = 0; mt < 2; ++mt) {
            const half_t* r = As + (wm * 32 + mt * 16 + nl) * 40;
            a[mt].h[0] = *(const v8h*)(r + 8 * g);
            a[mt].h[1] = *(const v8h*)(r + 16 + 8 * g);
        }
        #pragma unroll
        for (int nt = 0; nt < 2; ++nt) {
            AFrag b;
            const half_t* r = Bs + (wn * 32 + nt * 16 + nl) * 40;
            b.h[0] = *(const v8h*)(r + 8 * g);
            b.h[1] = *(const v8h*)(r + 16 + 8 * g);
            #pragma unroll
            for (int mt = 0; mt < 2; ++mt)
                acc[mt][nt] = wmma16(a[mt], b, acc[mt][nt]);
        }
        __syncthreads();
    }

    #pragma unroll
    for (int nt = 0; nt < 2; ++nt) {
        int n = nb + wn * 32 + nt * 16 + nl;
        float bv = bias[n];
        #pragma unroll
        for (int mt = 0; mt < 2; ++mt) {
            #pragma unroll
            for (int r = 0; r < 8; ++r) {
                int m = mb + wm * 32 + mt * 16 + r + 8 * g;
                float val = acc[mt][nt][r] + bv;
                if (mode == 0) {
                    int b = m >> 11, s = m & (SEQ - 1);
                    int h = n >> 6,  hd = n & (HD - 1);
                    ((half_t*)outp)[(((size_t)(b * NHEAD + h) * SEQ + s) << 6) + hd] =
                        (half_t)val;
                } else if (mode == 1) {
                    int b = m >> 11, s = m & (SEQ - 1);
                    int h = n >> 6,  hd = n & (HD - 1);
                    ((half_t*)outp)[((size_t)((b * NHEAD + h) * HD + hd) << 11) + s] =
                        (half_t)val;
                } else {
                    ((float*)outp)[(size_t)m * N + n] = val;
                }
            }
        }
    }
}

// -------------------------------------------------------- flash attention ---
// Q,K: [B,H,S,HD] f16 ; Vt: [B,H,HD,S] f16 ; O: [B,S,D] f16
__global__ __launch_bounds__(128) void flash_attn(
    const half_t* __restrict__ Q, const half_t* __restrict__ Km,
    const half_t* __restrict__ Vt, half_t* __restrict__ O,
    const int* __restrict__ causal_flag)
{
    __shared__ half_t P[4 * 16 * 40];   // per-wave 16x32 P tile, stride 40

    const int tid  = threadIdx.x, lane = tid & 31, wave = tid >> 5;
    const int g    = lane >> 4,   nl   = lane & 15;
    const int bh   = blockIdx.y;
    const int q0   = blockIdx.x * 64 + wave * 16;
    const int causal = *causal_flag;
    const size_t qk_base = (size_t)bh * SEQ * HD;
    const size_t vt_base = (size_t)bh * HD * SEQ;

    // Q fragments (hd 0..31 and 32..63)
    const half_t* qrow = Q + qk_base + (size_t)(q0 + nl) * HD;
    AFrag qa0, qa1;
    qa0.h[0] = *(const v8h*)(qrow + 8 * g);
    qa0.h[1] = *(const v8h*)(qrow + 16 + 8 * g);
    qa1.h[0] = *(const v8h*)(qrow + 32 + 8 * g);
    qa1.h[1] = *(const v8h*)(qrow + 48 + 8 * g);

    v8f o[4] = {}, z = {};
    float mrow[8], lrow[8];
    #pragma unroll
    for (int r = 0; r < 8; ++r) { mrow[r] = -1e30f; lrow[r] = 0.f; }

    const int nchunks = causal ? (blockIdx.x * 2 + 2) : (SEQ / 32);
    half_t* pl = P + wave * 16 * 40;

    for (int c = 0; c < nchunks; ++c) {
        const int key0 = c * 32;
        // scores for two 16-key tiles, K-dim = HD = 64 (two chained WMMAs each)
        AFrag kb;
        v8f s1 = z, s2 = z;
        const half_t* kr1 = Km + qk_base + (size_t)(key0 + nl) * HD;
        kb.h[0] = *(const v8h*)(kr1 + 8 * g);
        kb.h[1] = *(const v8h*)(kr1 + 16 + 8 * g);
        s1 = wmma16(qa0, kb, s1);
        kb.h[0] = *(const v8h*)(kr1 + 32 + 8 * g);
        kb.h[1] = *(const v8h*)(kr1 + 48 + 8 * g);
        s1 = wmma16(qa1, kb, s1);
        const half_t* kr2 = kr1 + 16 * HD;
        kb.h[0] = *(const v8h*)(kr2 + 8 * g);
        kb.h[1] = *(const v8h*)(kr2 + 16 + 8 * g);
        s2 = wmma16(qa0, kb, s2);
        kb.h[0] = *(const v8h*)(kr2 + 32 + 8 * g);
        kb.h[1] = *(const v8h*)(kr2 + 48 + 8 * g);
        s2 = wmma16(qa1, kb, s2);

        // online softmax, row = vgpr index r (+8g), 16 cols across lanes
        #pragma unroll
        for (int r = 0; r < 8; ++r) {
            float a  = s1[r] * 0.125f;
            float b2 = s2[r] * 0.125f;
            int   q  = q0 + r + 8 * g;
            if (causal) {
                if (key0 + nl > q)      a  = -1e30f;
                if (key0 + 16 + nl > q) b2 = -1e30f;
            }
            float tm = fmaxf(a, b2);
            tm = fmaxf(tm, __shfl_xor(tm, 1));
            tm = fmaxf(tm, __shfl_xor(tm, 2));
            tm = fmaxf(tm, __shfl_xor(tm, 4));
            tm = fmaxf(tm, __shfl_xor(tm, 8));
            float mn  = fmaxf(mrow[r], tm);
            float fac = __expf(mrow[r] - mn);
            float p1  = __expf(a - mn);
            float p2  = __expf(b2 - mn);
            float rs  = p1 + p2;
            rs += __shfl_xor(rs, 1);
            rs += __shfl_xor(rs, 2);
            rs += __shfl_xor(rs, 4);
            rs += __shfl_xor(rs, 8);
            lrow[r] = lrow[r] * fac + rs;
            mrow[r] = mn;
            o[0][r] *= fac; o[1][r] *= fac; o[2][r] *= fac; o[3][r] *= fac;
            int m = r + 8 * g;
            pl[m * 40 + nl]      = (half_t)p1;
            pl[m * 40 + 16 + nl] = (half_t)p2;
        }
        __syncthreads();

        // reload P as an A-fragment (16 q-rows x 32 keys)
        AFrag pa;
        const half_t* pr = pl + nl * 40;
        pa.h[0] = *(const v8h*)(pr + 8 * g);
        pa.h[1] = *(const v8h*)(pr + 16 + 8 * g);

        // O += P @ V   (N = HD = 64 -> 4 tiles; Vt rows are hd, contiguous keys)
        #pragma unroll
        for (int t = 0; t < 4; ++t) {
            AFrag vb;
            const half_t* vr = Vt + vt_base + (size_t)(t * 16 + nl) * SEQ
                               + key0 + 8 * g;
            vb.h[0] = *(const v8h*)(vr);
            vb.h[1] = *(const v8h*)(vr + 16);
            o[t] = wmma16(pa, vb, o[t]);
        }
        __syncthreads();
    }

    // normalize and write O in [B,S,D] (concat heads) layout
    const int b = bh >> 4, h = bh & 15;
    #pragma unroll
    for (int t = 0; t < 4; ++t) {
        #pragma unroll
        for (int r = 0; r < 8; ++r) {
            int   q   = q0 + r + 8 * g;
            float val = o[t][r] / lrow[r];
            O[((size_t)(b * SEQ + q)) * DIM + h * HD + t * 16 + nl] = (half_t)val;
        }
    }
}

// ------------------------------------------------------------------- host ---
extern "C" void kernel_launch(void* const* d_in, const int* in_sizes, int n_in,
                              void* d_out, int out_size, void* d_ws, size_t ws_size,
                              hipStream_t stream) {
    const float* x  = (const float*)d_in[0];
    const float* Wq = (const float*)d_in[1];
    const float* bq = (const float*)d_in[2];
    const float* Wk = (const float*)d_in[3];
    const float* bk = (const float*)d_in[4];
    const float* Wv = (const float*)d_in[5];
    const float* bv = (const float*)d_in[6];
    const float* Wo = (const float*)d_in[7];
    const float* bo = (const float*)d_in[8];
    const int* causal = (const int*)d_in[9];

    char* p = (char*)d_ws;
    half_t* xh  = (half_t*)p; p += (size_t)MTOT * DIM * 2;
    half_t* wqh = (half_t*)p; p += (size_t)DIM * DIM * 2;
    half_t* wkh = (half_t*)p; p += (size_t)DIM * DIM * 2;
    half_t* wvh = (half_t*)p; p += (size_t)DIM * DIM * 2;
    half_t* woh = (half_t*)p; p += (size_t)DIM * DIM * 2;
    half_t* Qh  = (half_t*)p; p += (size_t)MTOT * DIM * 2;
    half_t* Kh  = (half_t*)p; p += (size_t)MTOT * DIM * 2;
    half_t* Vth = (half_t*)p; p += (size_t)MTOT * DIM * 2;
    half_t* Oh  = (half_t*)p; p += (size_t)MTOT * DIM * 2;

    const int nx4 = MTOT * DIM / 4;     // 1M vec4
    const int nw4 = DIM * DIM / 4;      // 256K vec4
    cvt_f32_to_f16<<<(nx4 + 255) / 256, 256, 0, stream>>>(x,  xh,  nx4);
    cvt_f32_to_f16<<<(nw4 + 255) / 256, 256, 0, stream>>>(Wq, wqh, nw4);
    cvt_f32_to_f16<<<(nw4 + 255) / 256, 256, 0, stream>>>(Wk, wkh, nw4);
    cvt_f32_to_f16<<<(nw4 + 255) / 256, 256, 0, stream>>>(Wv, wvh, nw4);
    cvt_f32_to_f16<<<(nw4 + 255) / 256, 256, 0, stream>>>(Wo, woh, nw4);

    dim3 gblock(256);
    dim3 ggrid(DIM / 64, MTOT / 128);
    gemm_nt<<<ggrid, gblock, 0, stream>>>(xh, wqh, bq, Qh,  MTOT, DIM, DIM, 0);
    gemm_nt<<<ggrid, gblock, 0, stream>>>(xh, wkh, bk, Kh,  MTOT, DIM, DIM, 0);
    gemm_nt<<<ggrid, gblock, 0, stream>>>(xh, wvh, bv, Vth, MTOT, DIM, DIM, 1);

    flash_attn<<<dim3(SEQ / 64, BATCH * NHEAD), 128, 0, stream>>>(Qh, Kh, Vth, Oh, causal);

    gemm_nt<<<ggrid, gblock, 0, stream>>>(Oh, woh, bo, d_out, MTOT, DIM, DIM, 2);
}
